// CrossMamba_28879360099159
// MI455X (gfx1250) — matrix operand order repaired and checked by reading, not compiled, not run on hardware
//
#include <hip/hip_runtime.h>

typedef __attribute__((ext_vector_type(16))) _Float16 v16h;
typedef __attribute__((ext_vector_type(8)))  float    v8f;

#define BATCH  2
#define LSEQ   2048
#define DMODEL 1024
#define DSTATE 16
#define DCONV  4
#define DINNER 2048
#define DTRANK 64
#define RTOT   (DTRANK + 2 * DSTATE) /* 96 */

// ---------------------------------------------------------------------------
// WMMA helpers (CDNA5 v_wmma_f32_16x16x32_f16, wave32)
// A 16x32 f16 layout used here: row = lane&15, K = (lane>>4)*16 + 0..15
// B 32x16 f16 layout used here: col = lane&15, K = (lane>>4)*16 + 0..15
// C/D 16x16 f32: col = lane&15, rows = (lane>>4)*8 + j  (VGPR j)
// ---------------------------------------------------------------------------
__device__ __forceinline__ v8f wmma_f16(v16h a, v16h b, v8f c) {
  return __builtin_amdgcn_wmma_f32_16x16x32_f16(false, a, false, b, (short)0, c,
                                                false, false);
}

// A fragment from row-major f32 matrix S (ld = row stride), tile (row0, k0)
__device__ __forceinline__ v16h load_a_rowmajor(const float* __restrict__ S,
                                                int row0, int k0, int ld, int lane) {
  int row = row0 + (lane & 15);
  int kb  = k0 + ((lane >> 4) << 4);
  const float* p = S + (long)row * ld + kb;
  v16h a;
#pragma unroll
  for (int j = 0; j < 8; ++j) a[j]     = (_Float16)p[j];
#pragma unroll
  for (int j = 0; j < 8; ++j) a[8 + j] = (_Float16)p[8 + j];
  return a;
}

// B fragment where B[k][n] = S[n][k], S row-major (ld): contiguous per lane
__device__ __forceinline__ v16h load_bT(const float* __restrict__ S,
                                        int k0, int col0, int ld, int lane) {
  int col = col0 + (lane & 15);
  int kb  = k0 + ((lane >> 4) << 4);
  const float* p = S + (long)col * ld + kb;
  v16h b;
#pragma unroll
  for (int j = 0; j < 16; ++j) b[j] = (_Float16)p[j];
  return b;
}

// B fragment where B[k][n] = S[k][n], S row-major (ld): strided per lane
__device__ __forceinline__ v16h load_b_strided(const float* __restrict__ S,
                                               int k0, int col0, int ld, int lane) {
  int col = col0 + (lane & 15);
  int kb  = k0 + ((lane >> 4) << 4);
  const float* p = S + (long)kb * ld + col;
  v16h b;
#pragma unroll
  for (int j = 0; j < 16; ++j) b[j] = (_Float16)p[(long)j * ld];
  return b;
}

// Store C frag: element (m,n) -> dst[n*ldn + m]  (contiguous over the 8 m's)
__device__ __forceinline__ void store_c(float* __restrict__ dst, v8f c,
                                        int m0, int n0, long ldn, int lane) {
  int n = n0 + (lane & 15);
  int m = m0 + ((lane >> 4) << 3);
  float* p = dst + (long)n * ldn + m;
#pragma unroll
  for (int j = 0; j < 8; ++j) p[j] = c[j];
}

__device__ __forceinline__ float siluf(float v) {
  return v / (1.0f + __expf(-v));
}
__device__ __forceinline__ float softplusf(float x) {
  return (x > 20.0f) ? x : __logf(1.0f + __expf(x));
}

// ---------------------------------------------------------------------------
// K1: x = hs @ W1^T, z = in2 @ W2^T   (M=l, N=e, K=d) -> [b][e][l] layout
// ---------------------------------------------------------------------------
__global__ __launch_bounds__(32) void k_inproj(
    const float* __restrict__ hs, const float* __restrict__ in2,
    const float* __restrict__ w1, const float* __restrict__ w2,
    float* __restrict__ xraw, float* __restrict__ zraw) {
  int lane = threadIdx.x & 31;
  int l0 = blockIdx.x * 32;
  int e0 = blockIdx.y * 32;
  int b  = blockIdx.z >> 1;
  int which = blockIdx.z & 1;
  const float* A = (which ? in2 : hs) + (long)b * LSEQ * DMODEL; // [l][d]
  const float* W = which ? w2 : w1;                              // [e][d]
  float* dst = (which ? zraw : xraw) + (long)b * DINNER * LSEQ;  // [e][l]

  v8f c00{}, c01{}, c10{}, c11{};
  for (int k0 = 0; k0 < DMODEL; k0 += 32) {
    v16h a0 = load_a_rowmajor(A, l0,      k0, DMODEL, lane);
    v16h a1 = load_a_rowmajor(A, l0 + 16, k0, DMODEL, lane);
    v16h b0 = load_bT(W, k0, e0,      DMODEL, lane);
    v16h b1 = load_bT(W, k0, e0 + 16, DMODEL, lane);
    c00 = wmma_f16(a0, b0, c00);  c01 = wmma_f16(a0, b1, c01);
    c10 = wmma_f16(a1, b0, c10);  c11 = wmma_f16(a1, b1, c11);
  }
  store_c(dst, c00, l0,      e0,      LSEQ, lane);
  store_c(dst, c01, l0,      e0 + 16, LSEQ, lane);
  store_c(dst, c10, l0 + 16, e0,      LSEQ, lane);
  store_c(dst, c11, l0 + 16, e0 + 16, LSEQ, lane);
}

// ---------------------------------------------------------------------------
// K2: depthwise causal conv(+bias)+SiLU on x; SiLU on z; transpose to [b][l][e]
// ---------------------------------------------------------------------------
__global__ __launch_bounds__(256) void k_conv_silu_tr(
    const float* __restrict__ xraw, const float* __restrict__ zraw,
    const float* __restrict__ convw, const float* __restrict__ convb,
    float* __restrict__ xcT, float* __restrict__ zsT) {
  __shared__ float tile[32][33];
  int tid = threadIdx.x;
  int l0 = blockIdx.x * 32;
  int e0 = blockIdx.y * 32;
  int b  = blockIdx.z >> 1;
  int which = blockIdx.z & 1;
  const float* src = (which ? zraw : xraw) + (long)b * DINNER * LSEQ; // [e][l]
  float* dst = (which ? zsT : xcT) + (long)b * LSEQ * DINNER;         // [l][e]

  int ll = tid & 31;
  int eq = tid >> 5; // 0..7
#pragma unroll
  for (int ii = 0; ii < 4; ++ii) {
    int el = eq + ii * 8;
    int e  = e0 + el;
    int l  = l0 + ll;
    const float* row = src + (long)e * LSEQ;
    float v;
    if (which == 0) {
      float acc = convb[e];
#pragma unroll
      for (int k = 0; k < DCONV; ++k) {
        int li = l - (DCONV - 1) + k;
        float xv = (li >= 0) ? row[li] : 0.0f;
        acc += convw[e * DCONV + k] * xv;
      }
      v = acc;
    } else {
      v = row[l];
    }
    tile[el][ll] = siluf(v);
  }
  __syncthreads();
  int ew = tid & 31;
  int lq = tid >> 5;
#pragma unroll
  for (int ii = 0; ii < 4; ++ii) {
    int lw = lq + ii * 8;
    dst[(long)(l0 + lw) * DINNER + (e0 + ew)] = tile[ew][lw];
  }
}

// ---------------------------------------------------------------------------
// K3: x_dbl[b][r][l] = sum_e xcT[b][l][e] * xprojw[r][e]   (M=l, N=r, K=e)
// ---------------------------------------------------------------------------
__global__ __launch_bounds__(32) void k_xproj(
    const float* __restrict__ xcT, const float* __restrict__ xprojw,
    float* __restrict__ xdbl) {
  int lane = threadIdx.x & 31;
  int l0 = blockIdx.x * 32;
  int r0 = blockIdx.y * 32;
  int b  = blockIdx.z;
  const float* A = xcT + (long)b * LSEQ * DINNER; // [l][e]
  float* dst = xdbl + (long)b * RTOT * LSEQ;      // [r][l]
  v8f c00{}, c01{}, c10{}, c11{};
  for (int k0 = 0; k0 < DINNER; k0 += 32) {
    v16h a0 = load_a_rowmajor(A, l0,      k0, DINNER, lane);
    v16h a1 = load_a_rowmajor(A, l0 + 16, k0, DINNER, lane);
    v16h b0 = load_bT(xprojw, k0, r0,      DINNER, lane);
    v16h b1 = load_bT(xprojw, k0, r0 + 16, DINNER, lane);
    c00 = wmma_f16(a0, b0, c00);  c01 = wmma_f16(a0, b1, c01);
    c10 = wmma_f16(a1, b0, c10);  c11 = wmma_f16(a1, b1, c11);
  }
  store_c(dst, c00, l0,      r0,      LSEQ, lane);
  store_c(dst, c01, l0,      r0 + 16, LSEQ, lane);
  store_c(dst, c10, l0 + 16, r0,      LSEQ, lane);
  store_c(dst, c11, l0 + 16, r0 + 16, LSEQ, lane);
}

// ---------------------------------------------------------------------------
// K4: deltaT[b][l][e] = softplus(sum_r dt[b][r][l]*dtw[e][r] + dtb[e])
//     (M=e, N=l, K=r=64)
// ---------------------------------------------------------------------------
__device__ __forceinline__ void store_delta(float* __restrict__ dst, v8f c,
                                            int m0, int n0,
                                            const float* __restrict__ dtb, int lane) {
  int n = n0 + (lane & 15);
  int m = m0 + ((lane >> 4) << 3);
  float* p = dst + (long)n * DINNER + m;
#pragma unroll
  for (int j = 0; j < 8; ++j) p[j] = softplusf(c[j] + dtb[m + j]);
}

__global__ __launch_bounds__(32) void k_delta(
    const float* __restrict__ xdbl, const float* __restrict__ dtw,
    const float* __restrict__ dtb, float* __restrict__ deltaT) {
  int lane = threadIdx.x & 31;
  int l0 = blockIdx.x * 32;
  int e0 = blockIdx.y * 32;
  int b  = blockIdx.z;
  const float* Bs = xdbl + (long)b * RTOT * LSEQ; // rows 0..63 = dt, [r][l]
  float* dst = deltaT + (long)b * LSEQ * DINNER;
  v8f c00{}, c01{}, c10{}, c11{};
  for (int k0 = 0; k0 < DTRANK; k0 += 32) {
    v16h a0 = load_a_rowmajor(dtw, e0,      k0, DTRANK, lane);
    v16h a1 = load_a_rowmajor(dtw, e0 + 16, k0, DTRANK, lane);
    v16h b0 = load_b_strided(Bs, k0, l0,      LSEQ, lane);
    v16h b1 = load_b_strided(Bs, k0, l0 + 16, LSEQ, lane);
    c00 = wmma_f16(a0, b0, c00);  c01 = wmma_f16(a0, b1, c01);
    c10 = wmma_f16(a1, b0, c10);  c11 = wmma_f16(a1, b1, c11);
  }
  store_delta(dst, c00, e0,      l0,      dtb, lane);
  store_delta(dst, c01, e0,      l0 + 16, dtb, lane);
  store_delta(dst, c10, e0 + 16, l0,      dtb, lane);
  store_delta(dst, c11, e0 + 16, l0 + 16, dtb, lane);
}

// ---------------------------------------------------------------------------
// K5: selective scan, one lane per channel, states in registers.
//     yT[b][l][e] = (sum_n state_n * C_n + D[e]*u) * silu(z)
// ---------------------------------------------------------------------------
__global__ __launch_bounds__(32) void k_scan(
    const float* __restrict__ deltaT, const float* __restrict__ xcT,
    const float* __restrict__ zsT,   const float* __restrict__ xdbl,
    const float* __restrict__ A_log, const float* __restrict__ Dp,
    float* __restrict__ yT) {
  int lane = threadIdx.x & 31;
  int eg = blockIdx.x % (DINNER / 32);
  int b  = blockIdx.x / (DINNER / 32);
  int e  = eg * 32 + lane;
  float a[DSTATE], st[DSTATE];
#pragma unroll
  for (int n = 0; n < DSTATE; ++n) {
    a[n]  = -__expf(A_log[e * DSTATE + n]);
    st[n] = 0.0f;
  }
  float Dv = Dp[e];
  const float* bcbase = xdbl + ((long)b * RTOT + DTRANK) * LSEQ; // B rows, then C
  long base = (long)b * LSEQ * DINNER + e;
  for (int l = 0; l < LSEQ; ++l) {
    long off = base + (long)l * DINNER;
    float dlt = deltaT[off];
    float u   = xcT[off];
    float zs  = zsT[off];
    float du  = dlt * u;
    const float* bc = bcbase + l; // wave-uniform address
    float y = 0.0f;
#pragma unroll
    for (int n = 0; n < DSTATE; ++n) {
      float Bv = bc[(long)n * LSEQ];
      float Cv = bc[(long)(DSTATE + n) * LSEQ];
      float dA = __expf(dlt * a[n]);
      st[n] = st[n] * dA + du * Bv;
      y += st[n] * Cv;
    }
    yT[off] = (y + Dv * u) * zs;
  }
}

// ---------------------------------------------------------------------------
// K6: out[b][l][d] = sum_e outw[d][e] * yT[b][l][e]   (M=d, N=l, K=e)
// ---------------------------------------------------------------------------
__global__ __launch_bounds__(32) void k_outproj(
    const float* __restrict__ yT, const float* __restrict__ outw,
    float* __restrict__ out) {
  int lane = threadIdx.x & 31;
  int l0 = blockIdx.x * 32;
  int d0 = blockIdx.y * 32;
  int b  = blockIdx.z;
  const float* Bs = yT + (long)b * LSEQ * DINNER; // [l][e]
  float* dst = out + (long)b * LSEQ * DMODEL;     // [l][d]
  v8f c00{}, c01{}, c10{}, c11{};
  for (int k0 = 0; k0 < DINNER; k0 += 32) {
    v16h a0 = load_a_rowmajor(outw, d0,      k0, DINNER, lane);
    v16h a1 = load_a_rowmajor(outw, d0 + 16, k0, DINNER, lane);
    v16h b0 = load_bT(Bs, k0, l0,      DINNER, lane);
    v16h b1 = load_bT(Bs, k0, l0 + 16, DINNER, lane);
    c00 = wmma_f16(a0, b0, c00);  c01 = wmma_f16(a0, b1, c01);
    c10 = wmma_f16(a1, b0, c10);  c11 = wmma_f16(a1, b1, c11);
  }
  store_c(dst, c00, d0,      l0,      DMODEL, lane);
  store_c(dst, c01, d0,      l0 + 16, DMODEL, lane);
  store_c(dst, c10, d0 + 16, l0,      DMODEL, lane);
  store_c(dst, c11, d0 + 16, l0 + 16, DMODEL, lane);
}

// ---------------------------------------------------------------------------
extern "C" void kernel_launch(void* const* d_in, const int* in_sizes, int n_in,
                              void* d_out, int out_size, void* d_ws, size_t ws_size,
                              hipStream_t stream) {
  (void)in_sizes; (void)n_in; (void)out_size; (void)ws_size;
  const float* hs    = (const float*)d_in[0];
  const float* in2   = (const float*)d_in[1];
  const float* w1    = (const float*)d_in[2];
  const float* w2    = (const float*)d_in[3];
  const float* convw = (const float*)d_in[4];
  const float* convb = (const float*)d_in[5];
  const float* xprojw= (const float*)d_in[6];
  const float* dtw   = (const float*)d_in[7];
  const float* dtb   = (const float*)d_in[8];
  const float* A_log = (const float*)d_in[9];
  const float* Dp    = (const float*)d_in[10];
  const float* outw  = (const float*)d_in[11];

  const size_t S = (size_t)BATCH * DINNER * LSEQ * sizeof(float); // 33.55 MB
  char* ws = (char*)d_ws;
  float* xraw   = (float*)(ws + 0 * S);
  float* zraw   = (float*)(ws + 1 * S);
  float* xcT    = (float*)(ws + 2 * S);
  float* zsT    = (float*)(ws + 3 * S);
  float* xdbl   = (float*)(ws + 4 * S);           // 1.57 MB
  float* deltaT = zraw;  // zraw dead after K2
  float* yT     = xraw;  // xraw dead after K2

  dim3 g1(LSEQ / 32, DINNER / 32, 2 * BATCH);
  k_inproj<<<g1, 32, 0, stream>>>(hs, in2, w1, w2, xraw, zraw);

  dim3 g2(LSEQ / 32, DINNER / 32, 2 * BATCH);
  k_conv_silu_tr<<<g2, 256, 0, stream>>>(xraw, zraw, convw, convb, xcT, zsT);

  dim3 g3(LSEQ / 32, RTOT / 32, BATCH);
  k_xproj<<<g3, 32, 0, stream>>>(xcT, xprojw, xdbl);

  dim3 g4(LSEQ / 32, DINNER / 32, BATCH);
  k_delta<<<g4, 32, 0, stream>>>(xdbl, dtw, dtb, deltaT);

  k_scan<<<dim3(BATCH * DINNER / 32), 32, 0, stream>>>(deltaT, xcT, zsT, xdbl,
                                                       A_log, Dp, yT);

  dim3 g5(LSEQ / 32, DMODEL / 32, BATCH);
  k_outproj<<<g5, 32, 0, stream>>>(yT, outw, (float*)d_out);
}
